// InGramEntityLayer_52003464019982
// MI455X (gfx1250) — matrix operand order repaired and checked by reading, not compiled, not run on hardware
//
#include <hip/hip_runtime.h>

// ---------------------------------------------------------------------------
// InGram entity layer for MI455X (gfx1250, wave32, WMMA f16 16x16x32).
// ---------------------------------------------------------------------------

typedef __attribute__((ext_vector_type(16))) _Float16 v16h;
typedef __attribute__((ext_vector_type(8)))  float    v8f;

constexpr int NUM_ENT = 50000;
constexpr int NUM_REL = 64;
constexpr int NUM_TRI = 400000;
constexpr int DIN  = 128;
constexpr int DOUT = 128;
constexpr int DREL = 128;
constexpr int NH   = 8;
constexpr int NEDGE = NUM_TRI + NUM_ENT;   // 450000
constexpr int NTILE = NEDGE / 16;          // 28125 (exact)

static_assert(NUM_TRI % 16 == 0, "triple tiles must not straddle self edges");
static_assert(NEDGE % 16 == 0, "edge count must tile by 16");

// Orderable-uint mapping for float atomicMax (monotone, handles negatives).
__device__ __forceinline__ unsigned fenc(float f) {
    unsigned u = __float_as_uint(f);
    return (u & 0x80000000u) ? ~u : (u | 0x80000000u);
}
__device__ __forceinline__ float fdec(unsigned u) {
    return __uint_as_float((u & 0x80000000u) ? (u & 0x7FFFFFFFu) : ~u);
}

// ---------------------------------------------------------------------------
// P1: tail in-degree counts + per-tail relation-embedding sums (fp32 atomics).
// One triple handled by 32 threads; 4 floats each.
// ---------------------------------------------------------------------------
__global__ void k_count_relsum(const int* __restrict__ trip,
                               const float* __restrict__ emb_rel,
                               float* __restrict__ count,
                               float* __restrict__ relsum) {
    long long gid = (long long)blockIdx.x * blockDim.x + threadIdx.x;
    int t = (int)(gid >> 5);
    int c = (int)(gid & 31);
    if (t >= NUM_TRI) return;
    int rel  = trip[t * 3 + 1];
    int tail = trip[t * 3 + 2];
    if (c == 0) atomicAdd(&count[tail], 1.0f);
    const float* src = emb_rel + (size_t)rel * DREL + c * 4;
    float* dst = relsum + (size_t)tail * DREL + c * 4;
#pragma unroll
    for (int j = 0; j < 4; ++j) atomicAdd(&dst[j], src[j]);
}

// ---------------------------------------------------------------------------
// P2 helpers: fp32 -> f16 cast, self-rel divide, weight transpose to f16.
// ---------------------------------------------------------------------------
__global__ void k_cvt(const float* __restrict__ s, _Float16* __restrict__ d, int n) {
    int gid = blockIdx.x * blockDim.x + threadIdx.x;
    if (gid < n) d[gid] = (_Float16)s[gid];
}

__global__ void k_selfrel(const float* __restrict__ relsum,
                          const float* __restrict__ count,
                          _Float16* __restrict__ self16, int n) {
    int gid = blockIdx.x * blockDim.x + threadIdx.x;
    if (gid >= n) return;
    int e = gid >> 7;  // DREL == 128
    self16[gid] = (_Float16)(relsum[gid] / (count[e] + 1e-16f));
}

// W: K x N row-major fp32  ->  WT: N x K row-major f16 (column of W contiguous)
__global__ void k_transpose(const float* __restrict__ W, _Float16* __restrict__ WT,
                            int K, int N) {
    int gid = blockIdx.x * blockDim.x + threadIdx.x;
    if (gid >= K * N) return;
    int k = gid / N, n = gid % N;
    WT[(size_t)n * K + k] = (_Float16)W[gid];
}

// ---------------------------------------------------------------------------
// P3: attention logits.  One wave per 16-edge tile.
//   h = tanh([ent16[tail] | ent16[head] | rel16] @ Wa + ba)     (16 x 128)
//   attn_raw[e,h] = sum_d h[e, h*16+d] * attn_vec[h,d]
//   amax[tail,h]  = atomic max
// A-fragment (16x32 f16): lane L holds row M=L&15; halves j=0..7 at
// K = 32*kb + 8*(L>>4) + j, halves 8..15 at K+16 -> two contiguous 16B runs
// that never cross the 128-wide tail/head/rel feature segments.
// B-fragment (32x16 f16): lane L holds column N=L&15, same K order as A:
// 32*kb + 8*(L>>4) + j and +16 -> two 16B loads from the transposed weights.
// All 8 N-tile B-fragments per kb are loaded into distinct buffers first so
// the 8 WMMAs issue with graduated s_wait_loadcnt instead of full drains.
// ---------------------------------------------------------------------------
__global__ void __launch_bounds__(128)
k_attn(const int* __restrict__ trip,
       const _Float16* __restrict__ ent16,
       const _Float16* __restrict__ rel16,
       const _Float16* __restrict__ self16,
       const _Float16* __restrict__ WaT,     // 128 x 384 f16
       const float* __restrict__ ba,
       const float* __restrict__ attn_vec,   // 8 x 16
       float* __restrict__ attn,             // NEDGE x 8 (raw logits out)
       unsigned* __restrict__ amax) {        // NUM_ENT x 8 (orderable uint)
    int wave = (blockIdx.x * blockDim.x + threadIdx.x) >> 5;
    int lane = threadIdx.x & 31;
    if (wave >= NTILE) return;               // whole-wave uniform: EXEC stays all-ones

    int base = wave * 16;
    int m    = lane & 15;                    // A-matrix row owned by this lane
    int half = lane >> 4;                    // K sub-range selector
    int e    = base + m;

    const _Float16* tailrow;
    const _Float16* headrow;
    const _Float16* relrow;
    if (e < NUM_TRI) {
        int h = trip[e * 3 + 0], r = trip[e * 3 + 1], t = trip[e * 3 + 2];
        tailrow = ent16 + (size_t)t * DIN;
        headrow = ent16 + (size_t)h * DIN;
        relrow  = rel16 + (size_t)r * DREL;
    } else {
        int s = e - NUM_TRI;                 // self loop: head = tail = s
        tailrow = ent16 + (size_t)s * DIN;
        headrow = tailrow;
        relrow  = self16 + (size_t)s * DREL;
    }

    union Frag { v16h h; uint4 q[2]; };
    v8f acc[8] = {};
#pragma unroll
    for (int kb = 0; kb < 12; ++kb) {
        Frag A;
        const _Float16* segp = (kb < 4) ? tailrow : (kb < 8) ? headrow : relrow;
        int o0 = (kb & 3) * 32 + half * 8;
        A.q[0] = *(const uint4*)(segp + o0);
        A.q[1] = *(const uint4*)(segp + o0 + 16);
        Frag B[8];
#pragma unroll
        for (int t = 0; t < 8; ++t) {
            const _Float16* bp =
                WaT + (size_t)(t * 16 + (lane & 15)) * 384 + kb * 32 + half * 8;
            B[t].q[0] = *(const uint4*)(bp);
            B[t].q[1] = *(const uint4*)(bp + 16);
        }
#pragma unroll
        for (int t = 0; t < 8; ++t) {
            acc[t] = __builtin_amdgcn_wmma_f32_16x16x32_f16(
                false, A.h, false, B[t].h, (short)0, acc[t], false, false);
        }
    }

    // Epilogue: bias + tanh + attn_vec scale, reduce 16 columns per head.
    int n = lane & 15;                       // C/D column owned by this lane
#pragma unroll
    for (int t = 0; t < 8; ++t) {            // N-tile t == head t (DH == 16)
        float bav = ba[t * 16 + n];
        float av  = attn_vec[t * 16 + n];
#pragma unroll
        for (int r = 0; r < 8; ++r) {        // row M = r + 8*half
            float v = tanhf(acc[t][r] + bav) * av;
            v += __shfl_xor(v, 1);
            v += __shfl_xor(v, 2);
            v += __shfl_xor(v, 4);
            v += __shfl_xor(v, 8);           // stays within each 16-lane group
            if (n == 0) {
                int em = base + half * 8 + r;
                attn[(size_t)em * NH + t] = v;
                int tl = (em < NUM_TRI) ? trip[em * 3 + 2] : em - NUM_TRI;
                atomicMax(&amax[(size_t)tl * NH + t], fenc(v));
            }
        }
    }
}

// ---------------------------------------------------------------------------
// P4: attn_val = exp(raw - amax[tail]); accumulate segment sums.
// ---------------------------------------------------------------------------
__global__ void k_softmax_num(const int* __restrict__ trip,
                              const unsigned* __restrict__ amax,
                              float* __restrict__ attn,     // in: raw, out: exp
                              float* __restrict__ asum) {
    int gid = blockIdx.x * blockDim.x + threadIdx.x;
    if (gid >= NEDGE * NH) return;
    int e = gid >> 3, h = gid & 7;
    int tl = (e < NUM_TRI) ? trip[e * 3 + 2] : e - NUM_TRI;
    float mx = fdec(amax[(size_t)tl * NH + h]);
    float v = expf(attn[gid] - mx);
    attn[gid] = v;
    atomicAdd(&asum[(size_t)tl * NH + h], v);
}

// ---------------------------------------------------------------------------
// P5: messages.  msg = [ent16[head] | rel16] @ Wg + bg ;
//     out[tail] += beta * msg  (per-head softmax weight, fp32 atomics).
// ---------------------------------------------------------------------------
__global__ void __launch_bounds__(128)
k_msg(const int* __restrict__ trip,
      const _Float16* __restrict__ ent16,
      const _Float16* __restrict__ rel16,
      const _Float16* __restrict__ self16,
      const _Float16* __restrict__ WgT,      // 128 x 256 f16
      const float* __restrict__ bg,
      const float* __restrict__ attn,        // exp values
      const float* __restrict__ asum,
      float* __restrict__ out) {
    int wave = (blockIdx.x * blockDim.x + threadIdx.x) >> 5;
    int lane = threadIdx.x & 31;
    if (wave >= NTILE) return;

    int base = wave * 16;
    int m    = lane & 15;
    int half = lane >> 4;
    int e    = base + m;

    const _Float16* headrow;
    const _Float16* relrow;
    if (e < NUM_TRI) {
        int h = trip[e * 3 + 0], r = trip[e * 3 + 1];
        headrow = ent16 + (size_t)h * DIN;
        relrow  = rel16 + (size_t)r * DREL;
    } else {
        int s = e - NUM_TRI;
        headrow = ent16 + (size_t)s * DIN;
        relrow  = self16 + (size_t)s * DREL;
    }

    union Frag { v16h h; uint4 q[2]; };
    v8f acc[8] = {};
#pragma unroll
    for (int kb = 0; kb < 8; ++kb) {         // K = 256
        Frag A;
        const _Float16* segp = (kb < 4) ? headrow : relrow;
        int o0 = (kb & 3) * 32 + half * 8;
        A.q[0] = *(const uint4*)(segp + o0);
        A.q[1] = *(const uint4*)(segp + o0 + 16);
        Frag B[8];
#pragma unroll
        for (int t = 0; t < 8; ++t) {
            const _Float16* bp =
                WgT + (size_t)(t * 16 + (lane & 15)) * 256 + kb * 32 + half * 8;
            B[t].q[0] = *(const uint4*)(bp);
            B[t].q[1] = *(const uint4*)(bp + 16);
        }
#pragma unroll
        for (int t = 0; t < 8; ++t) {
            acc[t] = __builtin_amdgcn_wmma_f32_16x16x32_f16(
                false, A.h, false, B[t].h, (short)0, acc[t], false, false);
        }
    }

    // Tails for the 8 rows this lane's C-fragment covers.
    int tl[8];
#pragma unroll
    for (int r = 0; r < 8; ++r) {
        int em = base + half * 8 + r;
        tl[r] = (em < NUM_TRI) ? trip[em * 3 + 2] : em - NUM_TRI;
    }

    int n = lane & 15;
#pragma unroll
    for (int t = 0; t < 8; ++t) {            // head t, columns t*16 + n
        float bgv = bg[t * 16 + n];
#pragma unroll
        for (int r = 0; r < 8; ++r) {
            int em = base + half * 8 + r;
            float beta = attn[(size_t)em * NH + t] /
                         (asum[(size_t)tl[r] * NH + t] + 1e-16f);
            float val = beta * (acc[t][r] + bgv);
            atomicAdd(&out[(size_t)tl[r] * DOUT + t * 16 + n], val);
        }
    }
}

// ---------------------------------------------------------------------------
// Host launch
// ---------------------------------------------------------------------------
extern "C" void kernel_launch(void* const* d_in, const int* in_sizes, int n_in,
                              void* d_out, int out_size, void* d_ws, size_t ws_size,
                              hipStream_t stream) {
    const float* emb_ent  = (const float*)d_in[0];
    const float* emb_rel  = (const float*)d_in[1];
    const int*   trip     = (const int*)d_in[2];
    const float* Wa       = (const float*)d_in[3];
    const float* ba       = (const float*)d_in[4];
    const float* attn_vec = (const float*)d_in[5];
    const float* Wg       = (const float*)d_in[6];
    const float* bg       = (const float*)d_in[7];
    float* out = (float*)d_out;

    char* p = (char*)d_ws;
    auto alloc = [&](size_t bytes) -> void* {
        void* r = (void*)p;
        p += (bytes + 255) & ~(size_t)255;
        return r;
    };
    float*     count  = (float*)alloc((size_t)NUM_ENT * 4);
    float*     relsum = (float*)alloc((size_t)NUM_ENT * DREL * 4);
    float*     asum   = (float*)alloc((size_t)NUM_ENT * NH * 4);
    unsigned*  amax   = (unsigned*)alloc((size_t)NUM_ENT * NH * 4);
    float*     attn   = (float*)alloc((size_t)NEDGE * NH * 4);
    _Float16*  ent16  = (_Float16*)alloc((size_t)NUM_ENT * DIN * 2);
    _Float16*  rel16  = (_Float16*)alloc((size_t)NUM_REL * DREL * 2);
    _Float16*  self16 = (_Float16*)alloc((size_t)NUM_ENT * DREL * 2);
    _Float16*  WaT    = (_Float16*)alloc((size_t)384 * 128 * 2);
    _Float16*  WgT    = (_Float16*)alloc((size_t)256 * 128 * 2);

    hipMemsetAsync(count,  0, (size_t)NUM_ENT * 4, stream);
    hipMemsetAsync(relsum, 0, (size_t)NUM_ENT * DREL * 4, stream);
    hipMemsetAsync(asum,   0, (size_t)NUM_ENT * NH * 4, stream);
    hipMemsetAsync(amax,   0, (size_t)NUM_ENT * NH * 4, stream);  // 0 < enc(-inf): floor below every float
    hipMemsetAsync(out,    0, (size_t)NUM_ENT * DOUT * 4, stream);

    {   // P1
        long long n = (long long)NUM_TRI * 32;
        k_count_relsum<<<(unsigned)((n + 255) / 256), 256, 0, stream>>>(
            trip, emb_rel, count, relsum);
    }
    {   // P2
        int n = NUM_ENT * DIN;
        k_cvt<<<(n + 255) / 256, 256, 0, stream>>>(emb_ent, ent16, n);
        n = NUM_REL * DREL;
        k_cvt<<<(n + 255) / 256, 256, 0, stream>>>(emb_rel, rel16, n);
        n = NUM_ENT * DREL;
        k_selfrel<<<(n + 255) / 256, 256, 0, stream>>>(relsum, count, self16, n);
        n = 384 * 128;
        k_transpose<<<(n + 255) / 256, 256, 0, stream>>>(Wa, WaT, 384, 128);
        n = 256 * 128;
        k_transpose<<<(n + 255) / 256, 256, 0, stream>>>(Wg, WgT, 256, 128);
    }
    {   // P3: 4 waves / block, 1 tile / wave
        int blocks = (NTILE + 3) / 4;
        k_attn<<<blocks, 128, 0, stream>>>(trip, ent16, rel16, self16,
                                           WaT, ba, attn_vec, attn, amax);
    }
    {   // P4
        int n = NEDGE * NH;
        k_softmax_num<<<(n + 255) / 256, 256, 0, stream>>>(trip, amax, attn, asum);
    }
    {   // P5
        int blocks = (NTILE + 3) / 4;
        k_msg<<<blocks, 128, 0, stream>>>(trip, ent16, rel16, self16,
                                          WgT, bg, attn, asum, out);
    }
}
